// RegionAttention_56075093016885
// MI455X (gfx1250) — matrix operand order repaired and checked by reading, not compiled, not run on hardware
//
#include <hip/hip_runtime.h>
#include <hip/hip_bf16.h>

// Problem constants (from reference): B=2, C=512, H=W=64 -> N=4096, 8 heads, d=64
#define BATCH 2
#define CDIM 512
#define NTOK 4096
#define NHEAD 8
#define DHEAD 64
#define BCN ((size_t)BATCH * CDIM * NTOK)   // 4,194,304 elements per tensor

typedef __attribute__((ext_vector_type(16))) _Float16 v16h;
typedef __attribute__((ext_vector_type(4)))  _Float16 v4h;
typedef __attribute__((ext_vector_type(8)))  float    v8f;
typedef __attribute__((ext_vector_type(4)))  float    v4f;

// ---------------------------------------------------------------------------
// Fragment loaders for v_wmma_f32_16x16x32_f16 (wave32 layouts per CDNA5 ISA).
// Per-lane data is 16-byte contiguous -> each v16h lowers to 2x ds_load_b128.
//
// A (16x32, MxK), row-major source p[row*ld + k]:
//   lane&15 = row M; lanes 0-15 hold K {0..7,16..23}, lanes 16-31 {8..15,24..31}
// B (32x16, KxN) from K-contiguous source p[n*ld + k]:
//   lane&15 = col N; lanes 0-15 hold K {0..7,16..23}, lanes 16-31 {8..15,24..31}
// C/D (16x16): lane&15 = col N; VGPR j = row M=j (+8 for lanes 16-31).
// ---------------------------------------------------------------------------
__device__ __forceinline__ v16h load_a(const _Float16* p, int ld) {
  const int lane = threadIdx.x & 31;
  const int row  = lane & 15;
  const int kof  = (lane >> 4) ? 8 : 0;
  v16h a;
#pragma unroll
  for (int j = 0; j < 8; ++j) {
    const int k = ((j & 3) << 1) + ((j >> 2) << 4) + kof;  // {0..7,16..23}+kof
    a[2 * j]     = p[row * ld + k];
    a[2 * j + 1] = p[row * ld + k + 1];
  }
  return a;
}

// B fragment; source holds element (k,n) at p[n*ld + k] (K contiguous per col)
__device__ __forceinline__ v16h load_bk(const _Float16* p, int ld) {
  const int lane = threadIdx.x & 31;
  const int col  = lane & 15;
  const int kof  = (lane >> 4) ? 8 : 0;
  v16h b;
#pragma unroll
  for (int j = 0; j < 8; ++j) {
    const int k = ((j & 3) << 1) + ((j >> 2) << 4) + kof;  // {0..7,16..23}+kof
    b[2 * j]     = p[col * ld + k];
    b[2 * j + 1] = p[col * ld + k + 1];
  }
  return b;
}

__device__ __forceinline__ v8f wmma_f16(v16h a, v16h b, v8f c) {
  return __builtin_amdgcn_wmma_f32_16x16x32_f16(false, a, false, b, (short)0, c,
                                                false, false);
}

// ---------------------------------------------------------------------------
// Kernel 1: QKV projection. out[o,n] = Wqkv[o,:] . x[b,:,n] + bqkv[o]
// M=3C=1536, K=C=512, N=4096 per batch. Workgroup tile 64x64 with K-chunk 64
// (8 WMMA per barrier pair), 4 waves (2x2), wave tile 32x32.
// Epilogue scatters f16 results:
//   Q: [b][h][n][d]   (A-fragment friendly for flash)
//   K: [b][h][n][d]   (K^T B-fragments are d-contiguous)
//   V: [b][h][d][n]   (dim-major so P.V B-fragments are key-contiguous)
// ---------------------------------------------------------------------------
__global__ __launch_bounds__(128) void qkv_gemm(
    const float* __restrict__ x, const float* __restrict__ Wqkv,
    const float* __restrict__ bqkv, _Float16* __restrict__ qkv) {
  const int b   = blockIdx.z;
  const int n0  = blockIdx.x * 64;
  const int o0  = blockIdx.y * 64;
  const int tid = threadIdx.x;
  const int wid = tid >> 5, lane = tid & 31;
  const int hi  = lane >> 4, col = lane & 15;
  const int wm  = wid >> 1, wn = wid & 1;
  const int sr  = tid >> 4;         // staging row base (0..7)
  const int sc  = (tid & 15) * 4;   // staging col (x4 elements)

  __shared__ __align__(16) _Float16 At[64 * 72];  // W tile [64 o][64 c] pad->72
  __shared__ __align__(16) _Float16 Bt[64 * 72];  // x tile, transposed [64 n][64 c]

  v8f acc[2][2] = {};

  for (int kc = 0; kc < CDIM; kc += 64) {
    __syncthreads();
    // W tile: float4 global loads, packed 8B LDS stores
#pragma unroll
    for (int it = 0; it < 8; ++it) {
      const int r = sr + it * 8;
      const v4f f = *(const v4f*)&Wqkv[(size_t)(o0 + r) * CDIM + kc + sc];
      v4h hv;
      hv[0] = (_Float16)f[0]; hv[1] = (_Float16)f[1];
      hv[2] = (_Float16)f[2]; hv[3] = (_Float16)f[3];
      *(v4h*)&At[r * 72 + sc] = hv;
    }
    // x tile: float4 global loads, transposed scalar LDS stores
#pragma unroll
    for (int it = 0; it < 8; ++it) {
      const int r = sr + it * 8;  // channel (k)
      const v4f f = *(const v4f*)&x[((size_t)b * CDIM + kc + r) * NTOK + n0 + sc];
#pragma unroll
      for (int q = 0; q < 4; ++q) Bt[(sc + q) * 72 + r] = (_Float16)f[q];
    }
    __syncthreads();

    const v16h a0 = load_a(&At[(wm * 32) * 72], 72);        // k 0..31
    const v16h a1 = load_a(&At[(wm * 32 + 16) * 72], 72);
    const v16h a2 = load_a(&At[(wm * 32) * 72 + 32], 72);   // k 32..63
    const v16h a3 = load_a(&At[(wm * 32 + 16) * 72 + 32], 72);
    const v16h b0 = load_bk(&Bt[(wn * 32) * 72], 72);
    const v16h b1 = load_bk(&Bt[(wn * 32 + 16) * 72], 72);
    const v16h b2 = load_bk(&Bt[(wn * 32) * 72 + 32], 72);
    const v16h b3 = load_bk(&Bt[(wn * 32 + 16) * 72 + 32], 72);
    acc[0][0] = wmma_f16(a0, b0, acc[0][0]);
    acc[0][1] = wmma_f16(a0, b1, acc[0][1]);
    acc[1][0] = wmma_f16(a1, b0, acc[1][0]);
    acc[1][1] = wmma_f16(a1, b1, acc[1][1]);
    acc[0][0] = wmma_f16(a2, b2, acc[0][0]);
    acc[0][1] = wmma_f16(a2, b3, acc[0][1]);
    acc[1][0] = wmma_f16(a3, b2, acc[1][0]);
    acc[1][1] = wmma_f16(a3, b3, acc[1][1]);
  }

#pragma unroll
  for (int mi = 0; mi < 2; ++mi)
#pragma unroll
    for (int ni = 0; ni < 2; ++ni)
#pragma unroll
      for (int j = 0; j < 8; ++j) {
        const int o = o0 + wm * 32 + mi * 16 + hi * 8 + j;
        const int n = n0 + wn * 32 + ni * 16 + col;
        const float v = acc[mi][ni][j] + bqkv[o];
        const int which = o >> 9;       // 0=Q, 1=K, 2=V
        const int oc    = o & 511;
        const int head  = oc >> 6, dim = oc & 63;
        size_t addr;
        if (which == 2)   // V dim-major
          addr = 2 * BCN +
                 ((size_t)(b * NHEAD + head) * DHEAD + dim) * NTOK + n;
        else              // Q / K token-major
          addr = (size_t)which * BCN +
                 ((size_t)(b * NHEAD + head) * NTOK + n) * DHEAD + dim;
        qkv[addr] = (_Float16)v;
      }
}

// ---------------------------------------------------------------------------
// Kernel 2: flash attention, one block per (b, head, 64-query tile).
// 4 waves; wave w owns query rows [w*16, w*16+16). Key loop streams 64-key
// K/V tiles through LDS (vectorized copies); mask bias = (1-roi)*(-1e4) added
// to scaled scores; online softmax stats per-row in registers; P re-laid-out
// via LDS for the P.V WMMA. Output f16 token-major [b][n][c] for kernel 3.
// ---------------------------------------------------------------------------
__global__ __launch_bounds__(128) void flash_attn(
    const _Float16* __restrict__ qkv, const float* __restrict__ mask,
    _Float16* __restrict__ attn_out) {
  const int b  = blockIdx.z;
  const int h  = blockIdx.y;
  const int bh = b * NHEAD + h;
  const int q0 = blockIdx.x * 64;
  const int tid = threadIdx.x;
  const int wid = tid >> 5, lane = tid & 31;
  const int hi = lane >> 4, col = lane & 15;
  const int sr = tid >> 4;          // staging row base
  const int sc = (tid & 15) * 4;    // staging col (x4 halves = 8B)

  const _Float16* Q = qkv + ((size_t)bh * NTOK + q0) * DHEAD;    // [n][d]
  const _Float16* K = qkv + BCN + (size_t)bh * NTOK * DHEAD;     // [n][d]
  const _Float16* V = qkv + 2 * BCN + (size_t)bh * DHEAD * NTOK; // [d][n]
  const _Float16* Vrow = V + (size_t)sr * NTOK;                  // staging rows
  const _Float16* kpf  = K + (size_t)tid * 32;                   // prefetch base
  const _Float16* vpf  = V + (size_t)(tid >> 1) * NTOK;          // prefetch base

  __shared__ __align__(16) _Float16 Kt[64 * 72];  // [key][dim]
  __shared__ __align__(16) _Float16 Vt[64 * 72];  // [dim][key]
  __shared__ __align__(16) _Float16 Pt[64 * 72];  // [query][key]
  __shared__ float bias[64];

  // Q fragments for this wave's 16 rows, d split 0..31 / 32..63 (loaded once)
  const v16h Aq0 = load_a(Q + (size_t)wid * 16 * DHEAD, DHEAD);
  const v16h Aq1 = load_a(Q + (size_t)wid * 16 * DHEAD + 32, DHEAD);

  v8f   O[4] = {};
  float m[8], l[8];
#pragma unroll
  for (int j = 0; j < 8; ++j) { m[j] = -1e30f; l[j] = 0.0f; }

  for (int k0 = 0; k0 < NTOK; k0 += 64) {
    const _Float16* Kg = K + (size_t)k0 * DHEAD;  // 64 contiguous [key][dim] rows
    __syncthreads();
    // Stage K ([key][dim]) and V ([dim][key]) 64x64 tiles as 8-byte chunks
#pragma unroll
    for (int it = 0; it < 8; ++it) {
      const int r = sr + it * 8;
      *(uint2*)&Kt[r * 72 + sc] = *(const uint2*)&Kg[r * DHEAD + sc];
      *(uint2*)&Vt[r * 72 + sc] =
          *(const uint2*)&Vrow[(size_t)(it * 8) * NTOK + k0 + sc];
    }
    if (tid < 64)
      bias[tid] = (1.0f - mask[(size_t)b * NTOK + k0 + tid]) * (-10000.0f);
    if (k0 + 64 < NTOK) {
      __builtin_prefetch(kpf + (size_t)(k0 + 64) * DHEAD, 0, 1);
      __builtin_prefetch(vpf + k0 + 64, 0, 1);
    }
    __syncthreads();

    // S = (Q . K^T) * scale + bias   (4 column fragments of 16 keys)
    v8f S[4];
#pragma unroll
    for (int ni = 0; ni < 4; ++ni) {
      v8f s = {};
      s = wmma_f16(Aq0, load_bk(&Kt[(ni * 16) * 72], 72), s);       // d 0..31
      s = wmma_f16(Aq1, load_bk(&Kt[(ni * 16) * 72 + 32], 72), s);  // d 32..63
      S[ni] = s;
    }

    float mb[4];
#pragma unroll
    for (int ni = 0; ni < 4; ++ni) mb[ni] = bias[ni * 16 + col];

    // Online softmax per row (16-lane reductions stay inside lane half-group)
#pragma unroll
    for (int j = 0; j < 8; ++j) {
      float vmax = -1e30f;
#pragma unroll
      for (int ni = 0; ni < 4; ++ni) {
        const float s = S[ni][j] * 0.125f + mb[ni];
        S[ni][j] = s;
        vmax = fmaxf(vmax, s);
      }
#pragma unroll
      for (int off = 1; off < 16; off <<= 1)
        vmax = fmaxf(vmax, __shfl_xor(vmax, off, 32));
      const float mnew = fmaxf(m[j], vmax);

      float rs = 0.0f;
#pragma unroll
      for (int ni = 0; ni < 4; ++ni) {
        const float p = __expf(S[ni][j] - mnew);
        S[ni][j] = p;
        rs += p;
      }
#pragma unroll
      for (int off = 1; off < 16; off <<= 1) rs += __shfl_xor(rs, off, 32);

      const float alpha = __expf(m[j] - mnew);
      l[j] = l[j] * alpha + rs;
      m[j] = mnew;
#pragma unroll
      for (int ni = 0; ni < 4; ++ni) O[ni][j] *= alpha;
    }

    // Re-layout P (C-fragment regs -> A-fragment) via this wave's LDS rows
#pragma unroll
    for (int ni = 0; ni < 4; ++ni)
#pragma unroll
      for (int j = 0; j < 8; ++j)
        Pt[(wid * 16 + hi * 8 + j) * 72 + ni * 16 + col] = (_Float16)S[ni][j];

    // O += P . V   (keys split 0..31 / 32..63); V tile is [dim][key]
#pragma unroll
    for (int di = 0; di < 4; ++di) {
      O[di] = wmma_f16(load_a(&Pt[wid * 16 * 72], 72),
                       load_bk(&Vt[(di * 16) * 72], 72), O[di]);
      O[di] = wmma_f16(load_a(&Pt[wid * 16 * 72 + 32], 72),
                       load_bk(&Vt[(di * 16) * 72 + 32], 72), O[di]);
    }
  }

  // Normalize and store token-major f16: attn_out[b][n][h*64+dim]
#pragma unroll
  for (int di = 0; di < 4; ++di)
#pragma unroll
    for (int j = 0; j < 8; ++j) {
      const float o = O[di][j] / l[j];
      const int dim = di * 16 + col;
      const int n   = q0 + wid * 16 + hi * 8 + j;
      attn_out[((size_t)b * NTOK + n) * CDIM + h * DHEAD + dim] = (_Float16)o;
    }
}

// ---------------------------------------------------------------------------
// Kernel 3: output projection. out[b,o,n] = Wproj[o,:] . attn[b,n,:] + bproj[o]
// M=K=512, N=4096; token-major f16 B-matrix stages with pure vector copies.
// f32 output straight to d_out.
// ---------------------------------------------------------------------------
__global__ __launch_bounds__(128) void proj_gemm(
    const _Float16* __restrict__ ain, const float* __restrict__ Wproj,
    const float* __restrict__ bproj, float* __restrict__ out) {
  const int b   = blockIdx.z;
  const int n0  = blockIdx.x * 64;
  const int o0  = blockIdx.y * 64;
  const int tid = threadIdx.x;
  const int wid = tid >> 5, lane = tid & 31;
  const int hi  = lane >> 4, col = lane & 15;
  const int wm  = wid >> 1, wn = wid & 1;
  const int sr  = tid >> 4;
  const int sc  = (tid & 15) * 4;

  __shared__ __align__(16) _Float16 At[64 * 72];  // Wproj tile [64 o][64 c]
  __shared__ __align__(16) _Float16 Bt[64 * 72];  // attn tile [64 n][64 c]

  v8f acc[2][2] = {};

  for (int kc = 0; kc < CDIM; kc += 64) {
    __syncthreads();
#pragma unroll
    for (int it = 0; it < 8; ++it) {
      const int r = sr + it * 8;
      const v4f f = *(const v4f*)&Wproj[(size_t)(o0 + r) * CDIM + kc + sc];
      v4h hv;
      hv[0] = (_Float16)f[0]; hv[1] = (_Float16)f[1];
      hv[2] = (_Float16)f[2]; hv[3] = (_Float16)f[3];
      *(v4h*)&At[r * 72 + sc] = hv;
    }
#pragma unroll
    for (int it = 0; it < 8; ++it) {
      const int r = sr + it * 8;  // token row n
      *(uint2*)&Bt[r * 72 + sc] =
          *(const uint2*)&ain[((size_t)b * NTOK + n0 + r) * CDIM + kc + sc];
    }
    __syncthreads();

    const v16h a0 = load_a(&At[(wm * 32) * 72], 72);
    const v16h a1 = load_a(&At[(wm * 32 + 16) * 72], 72);
    const v16h a2 = load_a(&At[(wm * 32) * 72 + 32], 72);
    const v16h a3 = load_a(&At[(wm * 32 + 16) * 72 + 32], 72);
    const v16h b0 = load_bk(&Bt[(wn * 32) * 72], 72);
    const v16h b1 = load_bk(&Bt[(wn * 32 + 16) * 72], 72);
    const v16h b2 = load_bk(&Bt[(wn * 32) * 72 + 32], 72);
    const v16h b3 = load_bk(&Bt[(wn * 32 + 16) * 72 + 32], 72);
    acc[0][0] = wmma_f16(a0, b0, acc[0][0]);
    acc[0][1] = wmma_f16(a0, b1, acc[0][1]);
    acc[1][0] = wmma_f16(a1, b0, acc[1][0]);
    acc[1][1] = wmma_f16(a1, b1, acc[1][1]);
    acc[0][0] = wmma_f16(a2, b2, acc[0][0]);
    acc[0][1] = wmma_f16(a2, b3, acc[0][1]);
    acc[1][0] = wmma_f16(a3, b2, acc[1][0]);
    acc[1][1] = wmma_f16(a3, b3, acc[1][1]);
  }

#pragma unroll
  for (int mi = 0; mi < 2; ++mi)
#pragma unroll
    for (int ni = 0; ni < 2; ++ni)
#pragma unroll
      for (int j = 0; j < 8; ++j) {
        const int o = o0 + wm * 32 + mi * 16 + hi * 8 + j;
        const int n = n0 + wn * 32 + ni * 16 + col;
        out[((size_t)b * CDIM + o) * NTOK + n] = acc[mi][ni][j] + bproj[o];
      }
}

// ---------------------------------------------------------------------------
// Launcher. Workspace layout (f16): [0,BCN) Q token-major, [BCN,2*BCN) K
// token-major, [2*BCN,3*BCN) V dim-major, [3*BCN,4*BCN) attention output
// token-major.  Total = 32 MiB.
// ---------------------------------------------------------------------------
extern "C" void kernel_launch(void* const* d_in, const int* in_sizes, int n_in,
                              void* d_out, int out_size, void* d_ws,
                              size_t ws_size, hipStream_t stream) {
  const float* x     = (const float*)d_in[0];
  const float* mask  = (const float*)d_in[1];
  const float* Wqkv  = (const float*)d_in[2];
  const float* bqkv  = (const float*)d_in[3];
  const float* Wproj = (const float*)d_in[4];
  const float* bproj = (const float*)d_in[5];
  float* out = (float*)d_out;

  _Float16* qkv  = (_Float16*)d_ws;
  _Float16* attn = qkv + 3 * BCN;

  dim3 blk(128);
  qkv_gemm<<<dim3(NTOK / 64, (3 * CDIM) / 64, BATCH), blk, 0, stream>>>(
      x, Wqkv, bqkv, qkv);
  flash_attn<<<dim3(NTOK / 64, NHEAD, BATCH), blk, 0, stream>>>(qkv, mask,
                                                                attn);
  proj_gemm<<<dim3(NTOK / 64, CDIM / 64, BATCH), blk, 0, stream>>>(
      attn, Wproj, bproj, out);
}
